// MultiHeadAttention_71820443123791
// MI455X (gfx1250) — matrix operand order repaired and checked by reading, not compiled, not run on hardware
//
#include <hip/hip_runtime.h>
#include <hip/hip_bf16.h>
#include <math.h>

// ---------------------------------------------------------------------------
// MHA for gfx1250 (MI455X). All matmuls via v_wmma_f32_16x16x32_bf16.
// Problem constants (fixed by the reference):
namespace cfg {
constexpr int B  = 2;
constexpr int S  = 2048;   // Sq == Sk
constexpr int D  = 1024;
constexpr int H  = 16;
constexpr int DK = 64;     // D / H
}

typedef __bf16 bf16_t;
typedef __bf16 v16bf __attribute__((ext_vector_type(16)));
typedef float  v8f   __attribute__((ext_vector_type(8)));
typedef unsigned short ushort8 __attribute__((ext_vector_type(8)));

__device__ __forceinline__ unsigned short f2bf(float x) {
  union { float f; unsigned int u; } c; c.f = x;
  unsigned int u = c.u;
  u += 0x7FFFu + ((u >> 16) & 1u);          // round-to-nearest-even
  return (unsigned short)(u >> 16);
}

__device__ __forceinline__ v8f wmma_bf16(v16bf a, v16bf b, v8f c) {
  return __builtin_amdgcn_wmma_f32_16x16x32_bf16(
      /*neg_a=*/false, a, /*neg_b=*/false, b,
      /*c_mod=*/(short)0, c, /*reuse_a=*/false, /*reuse_b=*/false);
}

// A-fragment, 16x32 bf16 (ISA 7.12.2): lane L -> M = L&15;
// elems 0..7 : K = (L>>4)*8 + e ; elems 8..15 : K = 16 + (L>>4)*8 + (e-8)
__device__ __forceinline__ v16bf load_a_frag(const unsigned short* base, int ld) {
  const int lane = threadIdx.x & 31;
  const unsigned short* p = base + (size_t)(lane & 15) * ld + (lane >> 4) * 8;
  union { v16bf v; ushort8 h[2]; } f;
  f.h[0] = *(const ushort8*)(p);
  f.h[1] = *(const ushort8*)(p + 16);
  return f.v;
}

// Same but source rows are f32 (post-softmax scores in LDS). Pack pairs of
// floats to bf16 with one v_perm_b32 each (truncation; probs are in [0,1]).
__device__ __forceinline__ v16bf load_a_frag_f32(const float* base, int ld) {
  const int lane = threadIdx.x & 31;
  const float* p = base + (size_t)(lane & 15) * ld + (lane >> 4) * 8;
  const unsigned int* pu = (const unsigned int*)p;
  union { v16bf v; unsigned int w[8]; } f;
#pragma unroll
  for (int e = 0; e < 4; ++e) {
    f.w[e]     = __builtin_amdgcn_perm(pu[2 * e + 1],      pu[2 * e],      0x07060302u);
    f.w[4 + e] = __builtin_amdgcn_perm(pu[16 + 2 * e + 1], pu[16 + 2 * e], 0x07060302u);
  }
  return f.v;
}

// B-fragment, 32x16 (KxN) bf16: lane L -> N = L&15; elems 0..7 -> K = (L>>4)*8 + e; elems 8..15 -> K = 16 + (L>>4)*8 + (e-8).
// `base` points at (k0, n0) of a K-major-per-row [N][K] buffer with row stride ld.
__device__ __forceinline__ v16bf load_b_frag(const unsigned short* base, int ld) {
  const int lane = threadIdx.x & 31;
  const unsigned short* p = base + (size_t)(lane & 15) * ld + (lane >> 4) * 8;
  union { v16bf v; ushort8 h[2]; } f;
  f.h[0] = *(const ushort8*)(p);
  f.h[1] = *(const ushort8*)(p + 16);
  return f.v;
}

// ---------------------------------------------------------------------------
// GEMM: C[M,N] = A[M,K] @ W[N,K]^T + bias.  128x128 block, 8 waves, BK=32.
// A_BF16: A is bf16 (else f32). OUT_F32: write f32 (else bf16).
// VTRANS: scatter output into Vt[b][h][d][k] (head-transposed V for PV stage).
template <bool A_BF16, bool OUT_F32, bool VTRANS>
__global__ __launch_bounds__(256)
void gemm_bias_kernel(const void* __restrict__ Aptr, const float* __restrict__ W,
                      const float* __restrict__ bias, void* __restrict__ Out,
                      int M, int N, int K) {
  using namespace cfg;
  constexpr int BM = 128, BN = 128, BK = 32, LDT = 40;  // LDT pads banks, keeps 16B align
  __shared__ unsigned short As[BM * LDT];
  __shared__ unsigned short Bs[BN * LDT];

  const int t    = threadIdx.x;
  const int m0   = blockIdx.y * BM;
  const int n0   = blockIdx.x * BN;
  const int lrow = t >> 1;             // 0..127
  const int lcol = (t & 1) * 16;       // 0 or 16
  const int wave = t >> 5;
  const int lane = t & 31;
  const int half = lane >> 4, ln = lane & 15;
  const int wm = (wave >> 1) * 32;     // wave's M offset in block (4 groups)
  const int wn = (wave & 1) * 64;      // wave's N offset in block (2 groups)

  v8f acc[2][4] = {};

  for (int k0 = 0; k0 < K; k0 += BK) {
    // ---- stage A tile (BMxBK) as bf16 ----
    if constexpr (A_BF16) {
      const unsigned short* ga =
          (const unsigned short*)Aptr + (size_t)(m0 + lrow) * K + k0 + lcol;
      *(ushort8*)(As + lrow * LDT + lcol)     = *(const ushort8*)(ga);
      *(ushort8*)(As + lrow * LDT + lcol + 8) = *(const ushort8*)(ga + 8);
    } else {
      const float* ga = (const float*)Aptr + (size_t)(m0 + lrow) * K + k0 + lcol;
#pragma unroll
      for (int i = 0; i < 16; i += 4) {
        float4 v = *(const float4*)(ga + i);
        As[lrow * LDT + lcol + i + 0] = f2bf(v.x);
        As[lrow * LDT + lcol + i + 1] = f2bf(v.y);
        As[lrow * LDT + lcol + i + 2] = f2bf(v.z);
        As[lrow * LDT + lcol + i + 3] = f2bf(v.w);
      }
    }
    // ---- stage W tile (BNxBK) as bf16 ----
    {
      const float* gw = W + (size_t)(n0 + lrow) * K + k0 + lcol;
#pragma unroll
      for (int i = 0; i < 16; i += 4) {
        float4 v = *(const float4*)(gw + i);
        Bs[lrow * LDT + lcol + i + 0] = f2bf(v.x);
        Bs[lrow * LDT + lcol + i + 1] = f2bf(v.y);
        Bs[lrow * LDT + lcol + i + 2] = f2bf(v.z);
        Bs[lrow * LDT + lcol + i + 3] = f2bf(v.w);
      }
    }
    __syncthreads();

    v16bf a0 = load_a_frag(As + (size_t)wm * LDT, LDT);
    v16bf a1 = load_a_frag(As + (size_t)(wm + 16) * LDT, LDT);
#pragma unroll
    for (int nt = 0; nt < 4; ++nt) {
      v16bf bb = load_b_frag(Bs + (size_t)(wn + nt * 16) * LDT, LDT);
      acc[0][nt] = wmma_bf16(a0, bb, acc[0][nt]);
      acc[1][nt] = wmma_bf16(a1, bb, acc[1][nt]);
    }
    __syncthreads();
  }

  // ---- epilogue: bias add + store (C/D layout: n = lane&15, m = r + 8*(lane>>4)) ----
#pragma unroll
  for (int mt = 0; mt < 2; ++mt) {
#pragma unroll
    for (int nt = 0; nt < 4; ++nt) {
      const int gn = n0 + wn + nt * 16 + ln;
      const float bv = bias[gn];
#pragma unroll
      for (int r = 0; r < 8; ++r) {
        const int gm = m0 + wm + mt * 16 + half * 8 + r;
        const float val = acc[mt][nt][r] + bv;
        if constexpr (VTRANS) {
          const int bb = gm >> 11, sk = gm & (S - 1);
          const int hh = gn >> 6, dd = gn & (DK - 1);
          ((unsigned short*)Out)[((size_t)((bb * H + hh) * DK + dd)) * S + sk] = f2bf(val);
        } else if constexpr (OUT_F32) {
          ((float*)Out)[(size_t)gm * N + gn] = val;
        } else {
          ((unsigned short*)Out)[(size_t)gm * N + gn] = f2bf(val);
        }
      }
    }
  }
}

// ---------------------------------------------------------------------------
// Attention: one workgroup = 32 q rows x one (b,h). Full 32x2048 score stripe
// in LDS (f32), so the attn tensor is written to HBM exactly once.
// Masking is applied in the softmax max-pass where mask bytes are CONTIGUOUS
// per thread (coalesced uint4 loads) instead of scattered u8 loads per lane.
namespace attn_cfg {
constexpr int BQ  = 32;
constexpr int SLD = cfg::S + 8;   // padded f32 row stride (bank spread, 16B aligned)
constexpr int QLD = cfg::DK + 8;  // padded bf16 row stride
}

__global__ __launch_bounds__(256)
void attn_softmax_pv_kernel(const unsigned short* __restrict__ Qb,
                            const unsigned short* __restrict__ Kb,
                            const unsigned short* __restrict__ Vt,
                            const unsigned char* __restrict__ mask,
                            float* __restrict__ attnOut,
                            unsigned short* __restrict__ ctx) {
  using namespace cfg;
  using namespace attn_cfg;
  extern __shared__ char smem[];
  float*          Ssc  = (float*)smem;                                  // BQ x SLD
  unsigned short* Qs   = (unsigned short*)(smem + (size_t)BQ * SLD * 4);// BQ x QLD
  float*          redA = (float*)((char*)Qs + (size_t)BQ * QLD * 2);    // BQ x 8
  float*          redB = redA + BQ * 8;                                 // BQ x 8

  const int t    = threadIdx.x;
  const int q0   = blockIdx.x * BQ;
  const int h    = blockIdx.y;
  const int b    = blockIdx.z;
  const int lane = t & 31, wave = t >> 5;
  const int half = lane >> 4, ln = lane & 15;

  // ---- stage Q tile (32 x 64) ----
  {
    const int row = t >> 3, col = (t & 7) * 8;
    *(ushort8*)(Qs + row * QLD + col) =
        *(const ushort8*)(Qb + ((size_t)(b * S + q0 + row)) * D + h * DK + col);
  }
  __syncthreads();

  // Q A-fragments, cached in VGPRs, reused for all 16 k-tiles of this wave.
  v16bf aq[2][2];
#pragma unroll
  for (int mt = 0; mt < 2; ++mt)
#pragma unroll
    for (int kk = 0; kk < 2; ++kk)
      aq[mt][kk] = load_a_frag(Qs + (size_t)(mt * 16) * QLD + kk * 32, QLD);

  // ---- scores = Q K^T / 8 into LDS (no mask here; WMMA + ds_store only) ----
  {
    const int kbase = wave * 256;               // each wave owns 256 k columns
    const unsigned short* kcolbase = Kb + ((size_t)(b * S + kbase)) * D + h * DK;
    v16bf b0 = load_b_frag(kcolbase, D);        // d = 0..31
    v16bf b1 = load_b_frag(kcolbase + 32, D);   // d = 32..63
    for (int kt = 0; kt < 16; ++kt) {
      v16bf n0 = b0, n1 = b1;                   // prefetch next K fragments
      if (kt < 15) {
        const unsigned short* np = kcolbase + (size_t)(kt + 1) * 16 * D;
        n0 = load_b_frag(np, D);
        n1 = load_b_frag(np + 32, D);
      }
      const int kcol = kbase + kt * 16;
#pragma unroll
      for (int mt = 0; mt < 2; ++mt) {
        v8f acc = {};
        acc = wmma_bf16(aq[mt][0], b0, acc);
        acc = wmma_bf16(aq[mt][1], b1, acc);
        const int col = kcol + ln;
#pragma unroll
        for (int r = 0; r < 8; ++r) {
          const int row = mt * 16 + half * 8 + r;
          Ssc[(size_t)row * SLD + col] = acc[r] * 0.125f;   // 1/sqrt(dk)
        }
      }
      b0 = n0; b1 = n1;
    }
  }
  __syncthreads();

  // ---- mask + row softmax over 2048, 8 threads per row ----
  {
    const int row = t >> 3, seg = t & 7;
    float* rp = Ssc + (size_t)row * SLD + seg * 256;
    const unsigned char* mrow =
        mask + ((size_t)b * S + (q0 + row)) * S + seg * 256;

    float m = -INFINITY;
    auto apply4 = [&](unsigned int wv, int c) {
#pragma unroll
      for (int kk = 0; kk < 4; ++kk) {
        float v = rp[c + kk];
        if ((wv >> (8 * kk)) & 0xFFu) v = -INFINITY;   // where(mask, -inf, s)
        rp[c + kk] = v;
        m = fmaxf(m, v);
      }
    };
    for (int c0 = 0; c0 < 256; c0 += 16) {             // coalesced 16B mask loads
      const uint4 mb = *(const uint4*)(mrow + c0);
      apply4(mb.x, c0);
      apply4(mb.y, c0 + 4);
      apply4(mb.z, c0 + 8);
      apply4(mb.w, c0 + 12);
    }
    redA[row * 8 + seg] = m;
    __syncthreads();

    float rowmax = redA[row * 8];
#pragma unroll
    for (int i = 1; i < 8; ++i) rowmax = fmaxf(rowmax, redA[row * 8 + i]);
    float s = 0.f;
    for (int c = 0; c < 256; ++c) {
      float e = __expf(rp[c] - rowmax);
      rp[c] = e;
      s += e;
    }
    redB[row * 8 + seg] = s;
    __syncthreads();

    float rowsum = 0.f;
#pragma unroll
    for (int i = 0; i < 8; ++i) rowsum += redB[row * 8 + i];
    const float inv = 1.0f / rowsum;
    float* gp = attnOut + (((size_t)(b * H + h)) * S + (q0 + row)) * S + seg * 256;
    for (int c = 0; c < 256; c += 4) {          // normalize + single attn write
      float4 v = *(float4*)(rp + c);
      v.x *= inv; v.y *= inv; v.z *= inv; v.w *= inv;
      *(float4*)(rp + c) = v;
      *(float4*)(gp + c) = v;
    }
  }
  __syncthreads();

  // ---- out_h = attn @ V_h : each wave owns one 16x16 tile, full-K reduction ----
  {
    const int mt = wave >> 2;               // 0..1
    const int nt = (wave & 3) * 16;         // 0..48 (d offset)
    v8f acc = {};
    const unsigned short* vbase = Vt + (((size_t)(b * H + h)) * DK + nt) * (size_t)S;
    v16bf bb = load_b_frag(vbase, S);
    for (int ks = 0; ks < S; ks += 32) {
      v16bf nb = bb;                        // prefetch next V fragment
      if (ks + 32 < S) nb = load_b_frag(vbase + ks + 32, S);
      v16bf a = load_a_frag_f32(Ssc + (size_t)(mt * 16) * SLD + ks, SLD);
      acc = wmma_bf16(a, bb, acc);
      bb = nb;
    }
    const int d = nt + ln;
#pragma unroll
    for (int r = 0; r < 8; ++r) {
      const int row = q0 + mt * 16 + half * 8 + r;
      ctx[((size_t)(b * S) + row) * D + h * DK + d] = f2bf(acc[r]);
    }
  }
}

// ---------------------------------------------------------------------------
extern "C" void kernel_launch(void* const* d_in, const int* in_sizes, int n_in,
                              void* d_out, int out_size, void* d_ws, size_t ws_size,
                              hipStream_t stream) {
  using namespace cfg;
  const float*         x_q  = (const float*)d_in[0];
  const float*         x_kv = (const float*)d_in[1];
  const unsigned char* mask = (const unsigned char*)d_in[2];  // jax bool = 1 byte
  const float* wq = (const float*)d_in[3];
  const float* bq = (const float*)d_in[4];
  const float* wk = (const float*)d_in[5];
  const float* bk = (const float*)d_in[6];
  const float* wv = (const float*)d_in[7];
  const float* bv = (const float*)d_in[8];
  const float* wo = (const float*)d_in[9];
  const float* bo = (const float*)d_in[10];

  const size_t elems = (size_t)B * S * D;       // 4,194,304
  unsigned short* Qb  = (unsigned short*)d_ws;  // bf16 Q   (B,S,D)
  unsigned short* Kb  = Qb + elems;             // bf16 K   (B,S,D)
  unsigned short* Vt  = Kb + elems;             // bf16 V   (B,H,DK,S)  (d-major)
  unsigned short* ctx = Vt + elems;             // bf16 ctx (B,S,D)

  float* out   = (float*)d_out;                 // (B,S,D)
  float* attnG = out + elems;                   // (B,H,S,S)

  const int M = B * S, N = D, K = D;
  dim3 gg(N / 128, M / 128);                    // (8, 32)

  gemm_bias_kernel<false, false, false><<<gg, 256, 0, stream>>>(x_q,  wq, bq, Qb, M, N, K);
  gemm_bias_kernel<false, false, false><<<gg, 256, 0, stream>>>(x_kv, wk, bk, Kb, M, N, K);
  gemm_bias_kernel<false, false, true ><<<gg, 256, 0, stream>>>(x_kv, wv, bv, Vt, M, N, K);

  const size_t shmem = (size_t)attn_cfg::BQ * attn_cfg::SLD * 4 +
                       (size_t)attn_cfg::BQ * attn_cfg::QLD * 2 +
                       2 * (size_t)attn_cfg::BQ * 8 * 4;
  (void)hipFuncSetAttribute((const void*)attn_softmax_pv_kernel,
                            hipFuncAttributeMaxDynamicSharedMemorySize, (int)shmem);
  dim3 ga(S / attn_cfg::BQ, H, B);              // (64, 16, 2)
  attn_softmax_pv_kernel<<<ga, 256, shmem, stream>>>(Qb, Kb, Vt, mask, attnG, ctx);

  gemm_bias_kernel<true, true, false><<<gg, 256, 0, stream>>>(ctx, wo, bo, out, M, N, K);
}
